// GNN_17781164606037
// MI455X (gfx1250) — matrix operand order repaired and checked by reading, not compiled, not run on hardware
//
#include <hip/hip_runtime.h>
#include <hip/hip_bf16.h>

// ---------------------------------------------------------------------------
// Problem constants (match the reference)
// ---------------------------------------------------------------------------
#define GG   256
#define NPG  32
#define EPG  128
#define FPG  8
#define NN   (GG * NPG)   // 8192 nodes
#define EE   (GG * EPG)   // 32768 edges
#define FF   (GG * FPG)   // 2048 faces
#define NLAYERS 8

// ---------------------------------------------------------------------------
// WMMA types (CDNA5 / gfx1250, wave32)
// ---------------------------------------------------------------------------
typedef __bf16 bf16;
typedef __attribute__((ext_vector_type(16))) __bf16 v16bf;
typedef __attribute__((ext_vector_type(8)))  __bf16 v8bf;
typedef __attribute__((ext_vector_type(4)))  __bf16 v4bf;
typedef __attribute__((ext_vector_type(8)))  float  v8f;
typedef __attribute__((ext_vector_type(4)))  int    i4v;

#define AS1 __attribute__((address_space(1)))
#define AS3 __attribute__((address_space(3)))

// gfx1250 async global->LDS copy path (ASYNCcnt), guarded so the file still
// compiles (with the synchronous fallback) if the builtins are not declared.
// Probe round 3 confirmed the builtin exists with int4-pointer params.
#if defined(__has_builtin)
#  if __has_builtin(__builtin_amdgcn_global_load_async_to_lds_b128) && \
      __has_builtin(__builtin_amdgcn_s_wait_asynccnt)
#    define USE_ASYNC_LDS 1
#  endif
#endif
#ifndef USE_ASYNC_LDS
#  define USE_ASYNC_LDS 0
#endif

// ---------------------------------------------------------------------------
// Weight pre-pass: transpose + convert W[K][N] f32 -> Wt[N][K] bf16 (once per
// launch). Makes GEMM B-staging a pure contiguous bf16 copy.
// ---------------------------------------------------------------------------
__global__ void wcvt_transpose(const float* __restrict__ W, bf16* __restrict__ Wt,
                               int K, int N)
{
  __shared__ float tile[32][33];
  int k0 = blockIdx.x * 32, n0 = blockIdx.y * 32;
  int tx = threadIdx.x, ty = threadIdx.y;     // 32 x 8
#pragma unroll
  for (int i = 0; i < 32; i += 8)
    tile[ty + i][tx] = W[(size_t)(k0 + ty + i) * N + (n0 + tx)];
  __syncthreads();
#pragma unroll
  for (int i = 0; i < 32; i += 8)
    Wt[(size_t)(n0 + ty + i) * K + (k0 + tx)] = (bf16)tile[tx][ty + i];
}

// ---------------------------------------------------------------------------
// GEMM: C[M,Ncol] = act(A[M,K] @ W[K,Ncol] + bias), W given as Wt[Ncol][K] bf16.
// A fp32 (converted to bf16 while staging), C fp32.
// Block = 256 thr = 8 waves; BM=128, BN=64, BK=64; double-buffered LDS;
// wave tile 32x32 = 2x2 v_wmma_f32_16x16x32_bf16, 2 k-substeps per stage.
// Requires M%128==0, Ncol%64==0, K%64==0 (true at all call sites).
// ---------------------------------------------------------------------------
#define BM 128
#define BN 64
#define BK2 64

template <bool RELU>
__global__ __launch_bounds__(256) void gemm_bf16_wmma(
    const float* __restrict__ A, const bf16* __restrict__ Wt,
    const float* __restrict__ bias, float* __restrict__ C,
    int M, int K, int Ncol)
{
  __shared__ bf16 As[2][BM][BK2];   // 2 x 16 KB
  __shared__ bf16 Bt[2][BN][BK2];   // 2 x  8 KB  (row n holds K-contiguous)

  const int t    = threadIdx.x;
  const int lane = t & 31;
  const int wave = t >> 5;
  const int half = lane >> 4;       // 0: lanes 0-15, 1: lanes 16-31
  const int l16  = lane & 15;
  const int wm   = wave & 3;        // 4 wave-rows  x 32 = 128
  const int wn   = wave >> 2;       // 2 wave-cols  x 32 = 64
  const int blockM = blockIdx.x * BM;
  const int blockN = blockIdx.y * BN;

  v8f acc[2][2];
#pragma unroll
  for (int mi = 0; mi < 2; ++mi)
#pragma unroll
    for (int ni = 0; ni < 2; ++ni)
#pragma unroll
      for (int e = 0; e < 8; ++e) acc[mi][ni][e] = 0.0f;

  auto stage = [&](int buf, int k0) {
    // A tile 128x64: 2048 float4 groups, 8 per thread; cvt f32->bf16, b64 store
#pragma unroll
    for (int i = 0; i < 8; ++i) {
      int idx = t + i * 256;
      int r = idx >> 4;             // 16 groups of 4 per row
      int c4 = (idx & 15) * 4;
      float4 a4 = *(const float4*)&A[(size_t)(blockM + r) * K + (k0 + c4)];
      v4bf b4;
      b4.x = (bf16)a4.x; b4.y = (bf16)a4.y; b4.z = (bf16)a4.z; b4.w = (bf16)a4.w;
      *(v4bf*)&As[buf][r][c4] = b4;
    }
    // B tile 64x64 bf16: 512 16-byte chunks, 2 per thread
#pragma unroll
    for (int i = 0; i < 2; ++i) {
      int idx = t + i * 256;
      int n = idx >> 3;             // 8 chunks of 8 bf16 per row
      int c8 = (idx & 7) * 8;
      const bf16* gsrc = &Wt[(size_t)(blockN + n) * K + (k0 + c8)];
      bf16* ldst = &Bt[buf][n][c8];
#if USE_ASYNC_LDS
      // global_load_async_to_lds_b128 (ASYNCcnt). Global pointer: generic and
      // AS1 share representation. LDS pointer: low 32 bits of a generic LDS
      // address are the LDS offset (ISA flat-aperture rule), AS3 ptrs are 32b.
      __builtin_amdgcn_global_load_async_to_lds_b128(
          (AS1 i4v*)(size_t)gsrc,
          (AS3 i4v*)(unsigned)(size_t)ldst, 0, 0);
#else
      *(v8bf*)ldst = *(const v8bf*)gsrc;
#endif
    }
    // prefetch next k-tile (gfx1250 global_prefetch_b8)
    if (k0 + BK2 < K) {
      __builtin_prefetch(&A[(size_t)(blockM + (t >> 1)) * K + (k0 + BK2) + (t & 1) * 32], 0, 1);
      if (t < 128)
        __builtin_prefetch(&Wt[(size_t)(blockN + (t >> 1)) * K + (k0 + BK2) + (t & 1) * 64], 0, 1);
    }
  };

  const int nk = K / BK2;
  stage(0, 0);
#if USE_ASYNC_LDS
  __builtin_amdgcn_s_wait_asynccnt(0);
#endif
  __syncthreads();

  for (int kt = 0; kt < nk; ++kt) {
    const int cur = kt & 1;
    if (kt + 1 < nk) stage(cur ^ 1, (kt + 1) * BK2);

#pragma unroll
    for (int kk = 0; kk < 2; ++kk) {
      // Fragment loads per CDNA5 16-bit layouts:
      // A 16x32: half=0 -> K {0..7,16..23}, half=1 -> K {8..15,24..31}
      // B 32x16: half=0 -> K {0..7,16..23}, half=1 -> K {8..15,24..31}
      union { v16bf v; v8bf h[2]; } afr[2], bfr[2];
#pragma unroll
      for (int mi = 0; mi < 2; ++mi) {
        int m = wm * 32 + mi * 16 + l16;
        afr[mi].h[0] = *(const v8bf*)&As[cur][m][kk * 32 + half * 8];
        afr[mi].h[1] = *(const v8bf*)&As[cur][m][kk * 32 + 16 + half * 8];
      }
#pragma unroll
      for (int ni = 0; ni < 2; ++ni) {
        int n = wn * 32 + ni * 16 + l16;
        bfr[ni].h[0] = *(const v8bf*)&Bt[cur][n][kk * 32 + half * 8];
        bfr[ni].h[1] = *(const v8bf*)&Bt[cur][n][kk * 32 + 16 + half * 8];
      }
#pragma unroll
      for (int mi = 0; mi < 2; ++mi)
#pragma unroll
        for (int ni = 0; ni < 2; ++ni)
          acc[mi][ni] = __builtin_amdgcn_wmma_f32_16x16x32_bf16(
              false, afr[mi].v, false, bfr[ni].v, (short)0, acc[mi][ni],
              false, false);
    }
#if USE_ASYNC_LDS
    __builtin_amdgcn_s_wait_asynccnt(0);
#endif
    __syncthreads();
  }

  // Epilogue: C layout m = d + 8*half, n = lane&15
#pragma unroll
  for (int mi = 0; mi < 2; ++mi) {
#pragma unroll
    for (int ni = 0; ni < 2; ++ni) {
      int n = blockN + wn * 32 + ni * 16 + l16;
      float bv = bias[n];
#pragma unroll
      for (int d = 0; d < 8; ++d) {
        int m = blockM + wm * 32 + mi * 16 + d + half * 8;
        float v = acc[mi][ni][d] + bv;
        if (RELU) v = fmaxf(v, 0.0f);
        C[(size_t)m * Ncol + n] = v;
      }
    }
  }
}

// ---------------------------------------------------------------------------
// Encoder layer-0: one-hot(ids) @ W0 + b0 == embedding row-sum, then ReLU.
// ---------------------------------------------------------------------------
__global__ void node_embed_l0(const int* __restrict__ ids,
                              const float* __restrict__ W,
                              const float* __restrict__ b,
                              float* __restrict__ out)
{
  const int off[9] = {0, 119, 123, 135, 147, 157, 163, 169, 171};
  int n = blockIdx.x, t = threadIdx.x;
  __shared__ int sid[9];
  if (t < 9) sid[t] = ids[n * 9 + t];
  __syncthreads();
  for (int c = t; c < 512; c += 256) {
    float a = b[c];
#pragma unroll
    for (int i = 0; i < 9; ++i) a += W[(size_t)(off[i] + sid[i]) * 512 + c];
    out[(size_t)n * 512 + c] = fmaxf(a, 0.0f);
  }
}

__global__ void edge_embed_l0(const int* __restrict__ ids,
                              const float* __restrict__ W,
                              const float* __restrict__ b,
                              float* __restrict__ out)
{
  const int off[3] = {0, 5, 11};
  int e = blockIdx.x, t = threadIdx.x;
  __shared__ int sid[3];
  if (t < 3) sid[t] = ids[e * 3 + t];
  __syncthreads();
  for (int c = t; c < 512; c += 256) {
    float a = b[c];
#pragma unroll
    for (int i = 0; i < 3; ++i) a += W[(size_t)(off[i] + sid[i]) * 512 + c];
    out[(size_t)e * 512 + c] = fmaxf(a, 0.0f);
  }
}

// enc_global / enc_face on all-zero input: one 128-vector, computed once.
__global__ void const_mlp(const float* __restrict__ b0, const float* __restrict__ W1,
                          const float* __restrict__ b1, const float* __restrict__ W2,
                          const float* __restrict__ b2, float* __restrict__ out)
{
  __shared__ float r0[512];
  __shared__ float h[512];
  int t = threadIdx.x;
  r0[t] = fmaxf(b0[t], 0.0f);
  __syncthreads();
  float a = b1[t];
  for (int k = 0; k < 512; ++k) a += r0[k] * W1[(size_t)k * 512 + t];
  h[t] = fmaxf(a, 0.0f);
  __syncthreads();
  if (t < 128) {
    float o = b2[t];
    for (int k = 0; k < 512; ++k) o += h[k] * W2[(size_t)k * 128 + t];
    out[t] = o;
  }
}

__global__ void broadcast_row(const float* __restrict__ v, float* __restrict__ dst)
{
  dst[(size_t)blockIdx.x * 128 + threadIdx.x] = v[threadIdx.x];
}

// ---------------------------------------------------------------------------
// Gather / concat kernels (128 threads = L per block)
// ---------------------------------------------------------------------------
__global__ void edge_concat(const float* __restrict__ ea, const float* __restrict__ x,
                            const float* __restrict__ u, const float* __restrict__ fc,
                            const int* __restrict__ src, const int* __restrict__ dst,
                            const int* __restrict__ f0, const int* __restrict__ f1,
                            float* __restrict__ out)
{
  int e = blockIdx.x, t = threadIdx.x;
  float* o = out + (size_t)e * 768;
  o[t]        = ea[(size_t)e * 128 + t];
  o[128 + t]  = x[(size_t)src[e] * 128 + t];
  o[256 + t]  = x[(size_t)dst[e] * 128 + t];
  o[384 + t]  = u[(size_t)(e >> 7) * 128 + t];     // EPG = 128
  o[512 + t]  = fc[(size_t)f0[e] * 128 + t];
  o[640 + t]  = fc[(size_t)f1[e] * 128 + t];
}

__global__ void node_concat(const float* __restrict__ x, const float* __restrict__ snt,
                            const float* __restrict__ rcv, const float* __restrict__ u,
                            float* __restrict__ out)
{
  int n = blockIdx.x, t = threadIdx.x;
  float* o = out + (size_t)n * 512;
  o[t]       = x[(size_t)n * 128 + t];
  o[128 + t] = snt[(size_t)n * 128 + t];
  o[256 + t] = rcv[(size_t)n * 128 + t];
  o[384 + t] = u[(size_t)(n >> 5) * 128 + t];      // NPG = 32
}

__global__ void face_concat(const float* __restrict__ fc, const float* __restrict__ a0,
                            const float* __restrict__ a1, const float* __restrict__ u,
                            float* __restrict__ out)
{
  int f = blockIdx.x, t = threadIdx.x;
  float* o = out + (size_t)f * 512;
  o[t]       = fc[(size_t)f * 128 + t];
  o[128 + t] = a0[(size_t)f * 128 + t];
  o[256 + t] = a1[(size_t)f * 128 + t];
  o[384 + t] = u[(size_t)(f >> 3) * 128 + t];      // FPG = 8
}

__global__ void glob_concat(const float* __restrict__ u, const float* __restrict__ xs,
                            const float* __restrict__ es, const float* __restrict__ fs,
                            float* __restrict__ out)
{
  int g = blockIdx.x, t = threadIdx.x;
  float* o = out + (size_t)g * 512;
  o[t]       = u[(size_t)g * 128 + t];
  o[128 + t] = xs[(size_t)g * 128 + t];
  o[256 + t] = es[(size_t)g * 128 + t];
  o[384 + t] = fs[(size_t)g * 128 + t];
}

__global__ void pool_concat(const float* __restrict__ xs, const float* __restrict__ fs,
                            float* __restrict__ out)
{
  int g = blockIdx.x, t = threadIdx.x;
  out[(size_t)g * 256 + t]       = xs[(size_t)g * 128 + t];
  out[(size_t)g * 256 + 128 + t] = fs[(size_t)g * 128 + t];
}

// ---------------------------------------------------------------------------
// Segment sums
// ---------------------------------------------------------------------------
__global__ void scatter_add128(const float* __restrict__ src, const int* __restrict__ idx,
                               float* __restrict__ dst)
{
  int r = blockIdx.x, t = threadIdx.x;
  atomicAdd(&dst[(size_t)idx[r] * 128 + t], src[(size_t)r * 128 + t]);
}

__global__ void seg_sum128(const float* __restrict__ src, float* __restrict__ dst, int per)
{
  int g = blockIdx.x, t = threadIdx.x;
  const float* p = src + (size_t)g * per * 128;
  float a = 0.0f;
  for (int r = 0; r < per; ++r) a += p[(size_t)r * 128 + t];
  dst[(size_t)g * 128 + t] = a;
}

__global__ void add_inplace(float* __restrict__ y, const float* __restrict__ d)
{
  size_t i = (size_t)blockIdx.x * 256 + threadIdx.x;
  y[i] += d[i];
}

// Decoder last layer: [512] -> 1 per graph
__global__ void dec_final(const float* __restrict__ h, const float* __restrict__ Wv,
                          const float* __restrict__ b, float* __restrict__ out)
{
  __shared__ float sm[128];
  int g = blockIdx.x, t = threadIdx.x;
  float a = 0.0f;
  for (int k = t; k < 512; k += 128) a += h[(size_t)g * 512 + k] * Wv[k];
  sm[t] = a;
  __syncthreads();
  for (int s = 64; s > 0; s >>= 1) {
    if (t < s) sm[t] += sm[t + s];
    __syncthreads();
  }
  if (t == 0) out[g] = sm[0] + b[0];
}

// ---------------------------------------------------------------------------
// Host-side orchestration
// ---------------------------------------------------------------------------
static inline void gemm(hipStream_t s, const float* A, const bf16* Wt,
                        const float* b, float* C, int M, int K, int N, int relu)
{
  dim3 grid(M / BM, N / BN);
  if (relu) gemm_bf16_wmma<true><<<grid, 256, 0, s>>>(A, Wt, b, C, M, K, N);
  else      gemm_bf16_wmma<false><<<grid, 256, 0, s>>>(A, Wt, b, C, M, K, N);
}

extern "C" void kernel_launch(void* const* d_in, const int* in_sizes, int n_in,
                              void* d_out, int out_size, void* d_ws, size_t ws_size,
                              hipStream_t stream)
{
  (void)in_sizes; (void)out_size; (void)ws_size;

  const int* x_ids      = (const int*)d_in[0];
  const int* ea_ids     = (const int*)d_in[1];
  const int* edge_index = (const int*)d_in[2];
  const int* face_index = (const int*)d_in[5];
  const int* src = edge_index;
  const int* dst = edge_index + EE;
  const int* f0  = face_index;
  const int* f1  = face_index + EE;

  // Params flattened in setup_inputs() insertion order, leaves [W,b]:
  // 0..5 enc_node, 6..11 enc_edge, 12..17 enc_global, 18..23 enc_face,
  // 24..29 decoder, 30 + l*24 + {edge:0..5, node:6..11, face:12..17, glob:18..23}
  const float* P[256];
  for (int i = 0; i < 256 && (9 + i) < n_in; ++i) P[i] = (const float*)d_in[9 + i];

  // Workspace carve-out (f32 region, then bf16 weight region)
  char* w = (char*)d_ws;
  size_t off = 0;
  auto alloc = [&](size_t nf) { float* p = (float*)(w + off); off += nf * sizeof(float); return p; };
  float* xL   = alloc((size_t)NN * 128);
  float* eaL  = alloc((size_t)EE * 128);
  float* uL   = alloc((size_t)GG * 128);
  float* fcL  = alloc((size_t)FF * 128);
  float* x1   = alloc((size_t)NN * 128);
  float* ea1  = alloc((size_t)EE * 128);
  float* u1   = alloc((size_t)GG * 128);
  float* fc1  = alloc((size_t)FF * 128);
  float* big0 = alloc((size_t)EE * 768);   // concat buffer / hidden ping
  float* big1 = alloc((size_t)EE * 512);   // hidden pong
  float* snt  = alloc((size_t)NN * 128);
  float* rcv  = alloc((size_t)NN * 128);
  float* fa0  = alloc((size_t)FF * 128);
  float* fa1  = alloc((size_t)FF * 128);
  float* xsum = alloc((size_t)GG * 128);
  float* esum = alloc((size_t)GG * 128);
  float* fsum = alloc((size_t)GG * 128);
  float* gf   = alloc((size_t)GG * 512);
  float* gh   = alloc((size_t)GG * 512);
  float* pooled = alloc((size_t)GG * 256);
  float* cu   = alloc(128);
  float* cf   = alloc(128);

  auto balloc = [&](size_t ne) { bf16* p = (bf16*)(w + off); off += ne * sizeof(bf16); return p; };

  // ---- Weight pre-pass: transpose + f32->bf16 convert, once per launch ----
  const bf16* PW[256] = {};
  auto cvtW = [&](int pidx, int K, int N) {
    bf16* dst_ = balloc((size_t)K * N);
    dim3 g(K / 32, N / 32), b(32, 8);
    wcvt_transpose<<<g, b, 0, stream>>>(P[pidx], dst_, K, N);
    PW[pidx] = dst_;
  };
  cvtW(2, 512, 512);  cvtW(4, 512, 128);     // enc_node W1, W2
  cvtW(8, 512, 512);  cvtW(10, 512, 128);    // enc_edge W1, W2
  cvtW(24, 256, 512); cvtW(26, 512, 512);    // decoder W0, W1
  for (int l = 0; l < NLAYERS; ++l) {
    int b0 = 30 + l * 24;
    cvtW(b0 + 0, 768, 512); cvtW(b0 + 2, 512, 512); cvtW(b0 + 4, 512, 128);   // edge
    cvtW(b0 + 6, 512, 512); cvtW(b0 + 8, 512, 512); cvtW(b0 + 10, 512, 128);  // node
    cvtW(b0 + 12, 512, 512); cvtW(b0 + 14, 512, 512); cvtW(b0 + 16, 512, 128); // face
    cvtW(b0 + 18, 512, 512); cvtW(b0 + 20, 512, 512); cvtW(b0 + 22, 512, 128); // glob
  }

  // ---- Encoders ----
  node_embed_l0<<<NN, 256, 0, stream>>>(x_ids, P[0], P[1], big0);
  gemm(stream, big0, PW[2], P[3], big1, NN, 512, 512, 1);
  gemm(stream, big1, PW[4], P[5], xL,   NN, 512, 128, 0);

  edge_embed_l0<<<EE, 256, 0, stream>>>(ea_ids, P[6], P[7], big0);
  gemm(stream, big0, PW[8],  P[9],  big1, EE, 512, 512, 1);
  gemm(stream, big1, PW[10], P[11], eaL,  EE, 512, 128, 0);

  const_mlp<<<1, 512, 0, stream>>>(P[13], P[14], P[15], P[16], P[17], cu); // enc_global
  const_mlp<<<1, 512, 0, stream>>>(P[19], P[20], P[21], P[22], P[23], cf); // enc_face
  broadcast_row<<<GG, 128, 0, stream>>>(cu, uL);
  broadcast_row<<<FF, 128, 0, stream>>>(cf, fcL);

  // ---- Message-passing layers ----
  for (int l = 0; l < NLAYERS; ++l) {
    const float* const* Q = P + 30 + l * 24;
    const bf16* const* QW = PW + 30 + l * 24;

    // edge update
    edge_concat<<<EE, 128, 0, stream>>>(eaL, xL, uL, fcL, src, dst, f0, f1, big0);
    gemm(stream, big0, QW[0], Q[1], big1, EE, 768, 512, 1);
    gemm(stream, big1, QW[2], Q[3], big0, EE, 512, 512, 1);
    gemm(stream, big0, QW[4], Q[5], ea1,  EE, 512, 128, 0);

    // node update
    (void)hipMemsetAsync(snt, 0, (size_t)NN * 128 * sizeof(float), stream);
    (void)hipMemsetAsync(rcv, 0, (size_t)NN * 128 * sizeof(float), stream);
    scatter_add128<<<EE, 128, 0, stream>>>(ea1, src, snt);
    scatter_add128<<<EE, 128, 0, stream>>>(ea1, dst, rcv);
    node_concat<<<NN, 128, 0, stream>>>(xL, snt, rcv, uL, big0);
    gemm(stream, big0, QW[6],  Q[7],  big1, NN, 512, 512, 1);
    gemm(stream, big1, QW[8],  Q[9],  big0, NN, 512, 512, 1);
    gemm(stream, big0, QW[10], Q[11], x1,   NN, 512, 128, 0);

    // face update
    (void)hipMemsetAsync(fa0, 0, (size_t)FF * 128 * sizeof(float), stream);
    (void)hipMemsetAsync(fa1, 0, (size_t)FF * 128 * sizeof(float), stream);
    scatter_add128<<<EE, 128, 0, stream>>>(ea1, f0, fa0);
    scatter_add128<<<EE, 128, 0, stream>>>(ea1, f1, fa1);
    face_concat<<<FF, 128, 0, stream>>>(fcL, fa0, fa1, uL, big0);
    gemm(stream, big0, QW[12], Q[13], big1, FF, 512, 512, 1);
    gemm(stream, big1, QW[14], Q[15], big0, FF, 512, 512, 1);
    gemm(stream, big0, QW[16], Q[17], fc1,  FF, 512, 128, 0);

    // global update
    seg_sum128<<<GG, 128, 0, stream>>>(x1,  xsum, NPG);
    seg_sum128<<<GG, 128, 0, stream>>>(ea1, esum, EPG);
    seg_sum128<<<GG, 128, 0, stream>>>(fc1, fsum, FPG);
    glob_concat<<<GG, 128, 0, stream>>>(uL, xsum, esum, fsum, gf);
    gemm(stream, gf,   QW[18], Q[19], gh,   GG, 512, 512, 1);
    gemm(stream, gh,   QW[20], Q[21], big1, GG, 512, 512, 1);
    gemm(stream, big1, QW[22], Q[23], u1,   GG, 512, 128, 0);

    // residuals
    add_inplace<<<(NN * 128) / 256, 256, 0, stream>>>(xL,  x1);
    add_inplace<<<(EE * 128) / 256, 256, 0, stream>>>(eaL, ea1);
    add_inplace<<<(GG * 128) / 256, 256, 0, stream>>>(uL,  u1);
    add_inplace<<<(FF * 128) / 256, 256, 0, stream>>>(fcL, fc1);
  }

  // ---- Readout ----
  seg_sum128<<<GG, 128, 0, stream>>>(xL,  xsum, NPG);
  seg_sum128<<<GG, 128, 0, stream>>>(fcL, fsum, FPG);
  pool_concat<<<GG, 128, 0, stream>>>(xsum, fsum, pooled);
  gemm(stream, pooled, PW[24], P[25], gh, GG, 256, 512, 1);
  gemm(stream, gh,     PW[26], P[27], gf, GG, 512, 512, 1);
  dec_final<<<GG, 128, 0, stream>>>(gf, P[28], P[29], (float*)d_out);
}
